// CameraIPMEncoder_39101382263159
// MI455X (gfx1250) — compile-verified, matrix operand order checked by reading, not run on hardware
//
#include <hip/hip_runtime.h>
#include <math.h>
#include <stdint.h>

typedef _Float16 half_t;
typedef __attribute__((ext_vector_type(16))) _Float16 v16h;
typedef __attribute__((ext_vector_type(8)))  float    v8f;

typedef int v4i_t __attribute__((vector_size(16)));        // builtin's element type
typedef __attribute__((address_space(3))) v4i_t lds_v4i;   // ...in LDS address space

// gfx1250 async global->LDS copy path (ASYNCcnt), guarded for toolchain drift
#ifdef __has_builtin
#if __has_builtin(__builtin_amdgcn_global_load_async_to_lds_b128)
#define HAVE_ASYNC_LDS 1
#endif
#endif
#ifndef HAVE_ASYNC_LDS
#define HAVE_ASYNC_LDS 0
#endif

#if HAVE_ASYNC_LDS
#if __has_builtin(__builtin_amdgcn_s_wait_asynccnt)
#define ASYNC_WAIT() __builtin_amdgcn_s_wait_asynccnt(0)
#else
#define ASYNC_WAIT() asm volatile("s_wait_asynccnt 0x0" ::: "memory")
#endif
#endif

#define BM 128
#define BN 64
#define BK 32
#define LDSW 40   // halves per LDS row (80 bytes: 16B-aligned rows, staggered banks)

// ---------------------------------------------------------------------------
// Implicit-GEMM conv, NHWC f16 activations, [Co][Kpad] f16 weights,
// f32 accumulate via v_wmma_f32_16x16x32_f16, fused scale/bias (+res, +relu).
// Block = 128 threads = 4 waves; block tile 128x64; each wave owns a 64x32
// sub-tile => 8 WMMAs per K-step (4 A-frags x 2 B-frags).
// ---------------------------------------------------------------------------
__launch_bounds__(128)
__global__ void conv_wmma_kernel(
    const half_t* __restrict__ Ain, const half_t* __restrict__ Wk,
    const float* __restrict__ scale, const float* __restrict__ bias,
    const half_t* __restrict__ Res, half_t* __restrict__ Out,
    int NImg, int Hin, int Win, int Cin,
    int Ho, int Wo, int Co,
    int R, int S, int stride, int pad,
    int Ktot, int Kpad, int relu)
{
    __shared__ half_t lA[BM * LDSW];
    __shared__ half_t lB[BN * LDSW];

    const int M    = NImg * Ho * Wo;
    const int bm   = blockIdx.x * BM;
    const int bn   = blockIdx.y * BN;
    const int tid  = threadIdx.x;
    const int lane = tid & 31;
    const int wave = tid >> 5;
    const int wm   = (wave & 1) * 64;   // 2 waves tile M
    const int wn   = (wave >> 1) * 32;  // 2 waves tile N

    v8f acc[4][2] = {};

    // A loader: one output row per thread (32 halves of K per step)
    const int arow = tid;
    const int gm   = bm + arow;
    int img = 0, oy = 0, ox = 0;
    if (gm < M) { img = gm / (Ho * Wo); int r2 = gm - img * Ho * Wo; oy = r2 / Wo; ox = r2 - oy * Wo; }
    const int by   = oy * stride - pad;
    const int bx   = ox * stride - pad;
    const bool vrow = (gm < M);

    // B loader: 2 threads per weight row, 16 halves each
    const int bcol  = tid >> 1;
    const int bkoff = (tid & 1) * 16;
    const int gcol  = bn + bcol;

    const int nko = Kpad / BK;
    for (int ko = 0; ko < nko; ++ko) {
        const int k0 = ko * BK;

        // ---- A tile (im2col gather) ----
        {
            half_t* adst = &lA[arow * LDSW];
            if ((Cin & 31) == 0) {
                // 32-aligned K-chunk stays inside one (r,s) cell => contiguous NHWC
                int cell = k0 / Cin;
                int c    = k0 - cell * Cin;
                int r    = cell / S, s = cell - r * S;
                int iy   = by + r,  ix = bx + s;
                if (vrow && iy >= 0 && iy < Hin && ix >= 0 && ix < Win) {
                    const half_t* srcb =
                        &Ain[(((size_t)img * Hin + iy) * Win + ix) * Cin + c];
                    const float4* src = (const float4*)srcb;
#pragma unroll
                    for (int q = 0; q < 4; ++q) ((float4*)adst)[q] = src[q];
                    if (c + 32 < Cin)  // next K-chunk (common case): gfx1250 global_prefetch_b8
                        __builtin_prefetch(srcb + 32, 0, 1);
                } else {
#pragma unroll
                    for (int q = 0; q < 4; ++q)
                        ((float4*)adst)[q] = make_float4(0.f, 0.f, 0.f, 0.f);
                }
            } else {
                for (int e = 0; e < 32; ++e) {
                    int kk = k0 + e;
                    half_t v = (half_t)0.f;
                    if (vrow && kk < Ktot) {
                        int cell = kk / Cin;
                        int c    = kk - cell * Cin;
                        int r    = cell / S, s = cell - r * S;
                        int iy   = by + r,  ix = bx + s;
                        if (iy >= 0 && iy < Hin && ix >= 0 && ix < Win)
                            v = Ain[(((size_t)img * Hin + iy) * Win + ix) * Cin + c];
                    }
                    adst[e] = v;
                }
            }
        }
        // ---- B tile: weights [col][k], contiguous in k ----
        {
            half_t* bdst = &lB[bcol * LDSW + bkoff];
            if (gcol < Co) {
                const half_t* bsrc = &Wk[(size_t)gcol * Kpad + k0 + bkoff];
#if HAVE_ASYNC_LDS
                // flat LDS addr = {SHARED_BASE, offset[31:0]} -> low 32 bits are the LDS offset
                __builtin_amdgcn_global_load_async_to_lds_b128(
                    (v4i_t*)bsrc,
                    (lds_v4i*)(uint32_t)(uintptr_t)bdst, 0, 0);
                __builtin_amdgcn_global_load_async_to_lds_b128(
                    (v4i_t*)(bsrc + 8),
                    (lds_v4i*)(uint32_t)(uintptr_t)(bdst + 8), 0, 0);
#else
                ((float4*)bdst)[0] = ((const float4*)bsrc)[0];
                ((float4*)bdst)[1] = ((const float4*)bsrc)[1];
#endif
            } else {
                ((float4*)bdst)[0] = make_float4(0.f, 0.f, 0.f, 0.f);
                ((float4*)bdst)[1] = make_float4(0.f, 0.f, 0.f, 0.f);
            }
        }
#if HAVE_ASYNC_LDS
        ASYNC_WAIT();
#endif
        __syncthreads();

        // ---- fragment loads (ISA 16-bit layout: lanes 0-15 K0-7/16-23,
        //      lanes 16-31 K8-15/24-31) + 8 WMMAs per wave ----
        {
            const int rsel = lane & 15;
            const int ks   = (lane >> 4) * 8;
            v16h afrag[4], bfrag[2];
#pragma unroll
            for (int i = 0; i < 4; ++i) {
                union { float4 f[2]; v16h v; } ua;
                const half_t* pa = &lA[(wm + i * 16 + rsel) * LDSW];
                ua.f[0] = ((const float4*)(pa + ks))[0];
                ua.f[1] = ((const float4*)(pa + 16 + ks))[0];
                afrag[i] = ua.v;
            }
#pragma unroll
            for (int j = 0; j < 2; ++j) {
                union { float4 f[2]; v16h v; } ub;
                const half_t* pb = &lB[(wn + j * 16 + rsel) * LDSW];
                ub.f[0] = ((const float4*)(pb + ks))[0];
                ub.f[1] = ((const float4*)(pb + 16 + ks))[0];
                bfrag[j] = ub.v;
            }
#pragma unroll
            for (int i = 0; i < 4; ++i)
#pragma unroll
                for (int j = 0; j < 2; ++j)
                    acc[i][j] = __builtin_amdgcn_wmma_f32_16x16x32_f16(
                        false, afrag[i], false, bfrag[j],
                        (short)0, acc[i][j], false, false);
        }
        __syncthreads();
    }

    // ---- epilogue: C/D layout (vgpr v: row = v + 8*(lane>=16), col = lane&15) ----
    const int rl = lane & 15;
    const int rh = (lane >> 4) * 8;
#pragma unroll
    for (int i = 0; i < 4; ++i) {
#pragma unroll
        for (int j = 0; j < 2; ++j) {
            int gn = bn + wn + j * 16 + rl;
            if (gn >= Co) continue;
            float sc = scale[gn], bi = bias[gn];
#pragma unroll
            for (int v = 0; v < 8; ++v) {
                int gm2 = bm + wm + i * 16 + rh + v;
                if (gm2 >= M) continue;
                float f = acc[i][j][v] * sc + bi;
                if (Res) f += (float)Res[(size_t)gm2 * Co + gn];
                if (relu && f < 0.f) f = 0.f;
                Out[(size_t)gm2 * Co + gn] = (half_t)f;
            }
        }
    }
}

// ---------------------------------------------------------------------------
// Small helper kernels
// ---------------------------------------------------------------------------
__global__ void img_to_nhwc_f16(const float* __restrict__ in, half_t* __restrict__ out,
                                int NImg, int C, int H, int W)
{
    size_t idx = (size_t)blockIdx.x * blockDim.x + threadIdx.x;
    size_t tot = (size_t)NImg * H * W * C;
    if (idx >= tot) return;
    int c = idx % C; size_t r = idx / C;
    int x = r % W;  r /= W;
    int y = r % H;  int n = r / H;
    out[idx] = (half_t)in[(((size_t)n * C + c) * H + y) * W + x];
}

__global__ void maxpool3x3s2_nhwc(const half_t* __restrict__ in, half_t* __restrict__ out,
                                  int NImg, int Hin, int Win, int C, int Ho, int Wo)
{
    size_t idx = (size_t)blockIdx.x * blockDim.x + threadIdx.x;
    size_t tot = (size_t)NImg * Ho * Wo * C;
    if (idx >= tot) return;
    int c = idx % C; size_t r = idx / C;
    int ox = r % Wo; r /= Wo;
    int oy = r % Ho; int n = r / Ho;
    float m = -INFINITY;
    for (int ky = 0; ky < 3; ++ky) {
        int iy = oy * 2 - 1 + ky;
        if (iy < 0 || iy >= Hin) continue;
        for (int kx = 0; kx < 3; ++kx) {
            int ix = ox * 2 - 1 + kx;
            if (ix < 0 || ix >= Win) continue;
            float v = (float)in[(((size_t)n * Hin + iy) * Win + ix) * C + c];
            m = v > m ? v : m;
        }
    }
    out[idx] = (half_t)m;
}

// OIHW f32 -> [Co][Kpad] f16 with k = (r*S+s)*Ci + c, zero-padded past Ktot
__global__ void prep_weight(const float* __restrict__ w, half_t* __restrict__ wk,
                            int Co, int Ci, int R, int S, int Kpad)
{
    size_t idx = (size_t)blockIdx.x * blockDim.x + threadIdx.x;
    size_t tot = (size_t)Co * Kpad;
    if (idx >= tot) return;
    int o = idx / Kpad, k = idx % Kpad;
    int Ktot = R * S * Ci;
    half_t v = (half_t)0.f;
    if (k < Ktot) {
        int cell = k / Ci, c = k - cell * Ci;
        int r = cell / S, s = cell - r * S;
        v = (half_t)w[(((size_t)o * Ci + c) * R + r) * S + s];
    }
    wk[idx] = v;
}

// BN fold (+ optional conv bias): y = z*scale + bias'
__global__ void prep_bn(const float* __restrict__ g, const float* __restrict__ b,
                        const float* __restrict__ m, const float* __restrict__ v,
                        const float* __restrict__ cbias,
                        float* __restrict__ scale, float* __restrict__ bias, int C)
{
    int i = blockIdx.x * blockDim.x + threadIdx.x;
    if (i >= C) return;
    float s = g[i] * rsqrtf(v[i] + 1e-5f);
    float cb = cbias ? cbias[i] : 0.f;
    scale[i] = s;
    bias[i]  = (cb - m[i]) * s + b[i];
}

// ---------------------------------------------------------------------------
// IPM geometry: per feature-grid point compute BEV linear index + mask weight
// ---------------------------------------------------------------------------
__device__ inline void mat4_mul(const float* A, const float* B, float* C) {
    for (int i = 0; i < 4; ++i)
        for (int j = 0; j < 4; ++j) {
            float s = 0.f;
            for (int k = 0; k < 4; ++k) s += A[i * 4 + k] * B[k * 4 + j];
            C[i * 4 + j] = s;
        }
}
__device__ inline void inv_rigid4(const float* T, float* O) {
    for (int i = 0; i < 3; ++i)
        for (int j = 0; j < 3; ++j) O[i * 4 + j] = T[j * 4 + i];
    for (int i = 0; i < 3; ++i)
        O[i * 4 + 3] = -(T[0 * 4 + i] * T[3] + T[1 * 4 + i] * T[7] + T[2 * 4 + i] * T[11]);
    O[12] = 0.f; O[13] = 0.f; O[14] = 0.f; O[15] = 1.f;
}
__device__ inline void inv3(const float* K, float* O) {
    float a = K[0], b = K[1], c = K[2], d = K[3], e = K[4], f = K[5], g = K[6], h = K[7], i = K[8];
    float A = e * i - f * h, B = -(d * i - f * g), C = d * h - e * g;
    float inv = 1.f / (a * A + b * B + c * C);
    O[0] = A * inv;             O[1] = -(b * i - c * h) * inv; O[2] = (b * f - c * e) * inv;
    O[3] = B * inv;             O[4] = (a * i - c * g) * inv;  O[5] = -(a * f - c * d) * inv;
    O[6] = C * inv;             O[7] = -(a * h - b * g) * inv; O[8] = (a * e - b * d) * inv;
}

__global__ void geom_kernel(const float* __restrict__ cam2ego,
                            const float* __restrict__ cam_ego2global,
                            const float* __restrict__ intr,
                            const float* __restrict__ lidar2ego,
                            const float* __restrict__ ego2global,
                            int* __restrict__ lin, float* __restrict__ wmask,
                            int Ncam, int Hf, int Wf, float sx, float sy)
{
    int i = blockIdx.x * blockDim.x + threadIdx.x;
    int tot = Ncam * Hf * Wf;
    if (i >= tot) return;
    int n = i / (Hf * Wf);
    int r2 = i - n * Hf * Wf;
    int yy = r2 / Wf, xx = r2 - yy * Wf;

    float Li[16], Ei[16], Tlg[16], T1[16], Tlc[16];
    inv_rigid4(lidar2ego, Li);
    inv_rigid4(ego2global, Ei);
    mat4_mul(Li, Ei, Tlg);
    mat4_mul(Tlg, &cam_ego2global[n * 16], T1);
    mat4_mul(T1, &cam2ego[n * 16], Tlc);

    float Kinv[9];
    inv3(&intr[n * 9], Kinv);

    float px = ((float)xx + 0.5f) * sx;
    float py = ((float)yy + 0.5f) * sy;
    float dc[3];
    for (int j = 0; j < 3; ++j)
        dc[j] = Kinv[j * 3 + 0] * px + Kinv[j * 3 + 1] * py + Kinv[j * 3 + 2];
    float nrm = sqrtf(dc[0] * dc[0] + dc[1] * dc[1] + dc[2] * dc[2]);
    nrm = nrm < 1e-6f ? 1e-6f : nrm;
    dc[0] /= nrm; dc[1] /= nrm; dc[2] /= nrm;

    float dl[3];
    for (int j = 0; j < 3; ++j)
        dl[j] = Tlc[j * 4 + 0] * dc[0] + Tlc[j * 4 + 1] * dc[1] + Tlc[j * 4 + 2] * dc[2];
    float tx = Tlc[3], ty = Tlc[7], tz = Tlc[11];

    bool mask = fabsf(dl[2]) > 1e-3f;
    float tt = mask ? (-tz / dl[2]) : 0.f;
    mask = mask && (tt > 0.f);
    float wx = tx + dl[0] * tt;
    float wy = ty + dl[1] * tt;
    mask = mask && (wx >= -50.f) && (wx <= 50.f) && (wy >= -50.f) && (wy <= 50.f);
    int ix = (int)floorf((wx + 50.f) / 0.5f);
    int iy = (int)floorf((wy + 50.f) / 0.5f);
    ix = ix < 0 ? 0 : (ix > 199 ? 199 : ix);
    iy = iy < 0 ? 0 : (iy > 199 ? 199 : iy);
    lin[i]   = mask ? (iy * 200 + ix) : 0;
    wmask[i] = mask ? 1.f : 0.f;
}

__global__ void zero_f32(float* __restrict__ p, size_t n)
{
    size_t i = (size_t)blockIdx.x * blockDim.x + threadIdx.x;
    if (i < n) p[i] = 0.f;
}

__global__ void scatter_bev(const half_t* __restrict__ feats, const int* __restrict__ lin,
                            const float* __restrict__ wmask, float* __restrict__ out,
                            int npts, int C)
{
    int idx = blockIdx.x * blockDim.x + threadIdx.x;
    if (idx >= npts * C) return;
    int pt = idx / C, c = idx - pt * C;
    float w = wmask[pt];
    if (w == 0.f) return;
    float v = (float)feats[(size_t)pt * C + c] * w;
    atomicAdd(&out[(size_t)c * 40000 + lin[pt]], v);
}

// ---------------------------------------------------------------------------
// Host orchestration
// ---------------------------------------------------------------------------
extern "C" void kernel_launch(void* const* d_in, const int* in_sizes, int n_in,
                              void* d_out, int out_size, void* d_ws, size_t ws_size,
                              hipStream_t stream)
{
    (void)in_sizes; (void)n_in; (void)ws_size;
    const float* images         = (const float*)d_in[0];
    const float* cam2ego        = (const float*)d_in[1];
    const float* cam_ego2global = (const float*)d_in[2];
    const float* intrinsics     = (const float*)d_in[3];
    const float* lidar2ego      = (const float*)d_in[4];
    const float* ego2global     = (const float*)d_in[5];

    // ---- flatten param indices (dict insertion order; bn tuples = 4 arrays) ----
    struct PSpec { int wi, bn, cb, Co, Ci, R, S; };
    PSpec sp[21];
    int ns = 0, p = 6;
    { int w = p++; int bn = p; p += 4; sp[ns++] = {w, bn, -1, 64, 3, 7, 7}; }
    int couts[4] = {64, 128, 256, 512}, strides[4] = {1, 2, 2, 2};
    int cin = 64;
    for (int l = 0; l < 4; ++l) {
        int cout = couts[l];
        for (int b = 0; b < 2; ++b) {
            int ci = (b == 0) ? cin : cout;
            int st = (b == 0) ? strides[l] : 1;
            int w1 = p++; int bn1 = p; p += 4;
            int w2 = p++; int bn2 = p; p += 4;
            sp[ns++] = {w1, bn1, -1, cout, ci, 3, 3};
            sp[ns++] = {w2, bn2, -1, cout, cout, 3, 3};
            if (b == 0 && (st != 1 || ci != cout)) {
                int dw = p++; int dbn = p; p += 4;
                sp[ns++] = {dw, dbn, -1, cout, ci, 1, 1};
            }
        }
        cin = cout;
    }
    { int hw = p++; int hb = p++; int hbn = p; p += 4;
      sp[ns++] = {hw, hbn, hb, 64, 512, 1, 1}; }

    // ---- workspace bump allocator ----
    char* ws = (char*)d_ws;
    size_t off = 0;
    auto alloc = [&](size_t bytes) -> void* {
        size_t a = (off + 255) & ~(size_t)255;
        off = a + bytes;
        return (void*)(ws + a);
    };
    auto cdiv = [](long a, long b) { return (a + b - 1) / b; };

    // ---- prep all conv weights / BN folds ----
    half_t* wkp[21]; float* scp[21]; float* bip[21]; int kpad[21];
    for (int s = 0; s < ns; ++s) {
        int Ktot = sp[s].R * sp[s].S * sp[s].Ci;
        int Kp = (Ktot + 31) & ~31;
        kpad[s] = Kp;
        wkp[s] = (half_t*)alloc((size_t)sp[s].Co * Kp * sizeof(half_t));
        scp[s] = (float*)alloc(sp[s].Co * sizeof(float));
        bip[s] = (float*)alloc(sp[s].Co * sizeof(float));
        size_t tot = (size_t)sp[s].Co * Kp;
        prep_weight<<<cdiv(tot, 256), 256, 0, stream>>>(
            (const float*)d_in[sp[s].wi], wkp[s], sp[s].Co, sp[s].Ci, sp[s].R, sp[s].S, Kp);
        const float* cb = sp[s].cb >= 0 ? (const float*)d_in[sp[s].cb] : nullptr;
        prep_bn<<<cdiv(sp[s].Co, 128), 128, 0, stream>>>(
            (const float*)d_in[sp[s].bn + 0], (const float*)d_in[sp[s].bn + 1],
            (const float*)d_in[sp[s].bn + 2], (const float*)d_in[sp[s].bn + 3],
            cb, scp[s], bip[s], sp[s].Co);
    }

    // ---- activation buffers (NHWC f16) ----
    const int NC = 6;
    half_t* IMG  = (half_t*)alloc((size_t)NC * 256 * 704 * 3 * sizeof(half_t));
    half_t* ABIG = (half_t*)alloc((size_t)NC * 128 * 352 * 64 * sizeof(half_t));
    size_t mbytes = (size_t)NC * 64 * 176 * 64 * sizeof(half_t);
    half_t* M0 = (half_t*)alloc(mbytes);
    half_t* M1 = (half_t*)alloc(mbytes);
    half_t* M2 = (half_t*)alloc(mbytes);
    half_t* M3 = (half_t*)alloc(mbytes);
    const int NPTS = NC * 8 * 22;
    int*   LIN = (int*)alloc(NPTS * sizeof(int));
    float* WMK = (float*)alloc(NPTS * sizeof(float));

    auto conv = [&](int s, const half_t* A, const half_t* Resid, half_t* Outp,
                    int Hin, int Win, int Ho, int Wo, int st, int pad, int relu) {
        int M = NC * Ho * Wo;
        dim3 grid((unsigned)cdiv(M, BM), (unsigned)cdiv(sp[s].Co, BN));
        conv_wmma_kernel<<<grid, 128, 0, stream>>>(
            A, wkp[s], scp[s], bip[s], Resid, Outp,
            NC, Hin, Win, sp[s].Ci, Ho, Wo, sp[s].Co,
            sp[s].R, sp[s].S, st, pad,
            sp[s].R * sp[s].S * sp[s].Ci, kpad[s], relu);
    };

    // ---- pipeline ----
    {
        size_t tot = (size_t)NC * 256 * 704 * 3;
        img_to_nhwc_f16<<<cdiv(tot, 256), 256, 0, stream>>>(images, IMG, NC, 3, 256, 704);
    }
    conv(0, IMG, nullptr, ABIG, 256, 704, 128, 352, 2, 3, 1);           // conv1 7x7/2
    {
        size_t tot = (size_t)NC * 64 * 176 * 64;
        maxpool3x3s2_nhwc<<<cdiv(tot, 256), 256, 0, stream>>>(ABIG, M0, NC, 128, 352, 64, 64, 176);
    }
    // layer1 @ 64x176, 64ch (specs 1..4)
    conv(1, M0, nullptr, M1, 64, 176, 64, 176, 1, 1, 1);
    conv(2, M1, M0,      M2, 64, 176, 64, 176, 1, 1, 1);
    conv(3, M2, nullptr, M1, 64, 176, 64, 176, 1, 1, 1);
    conv(4, M1, M2,      M0, 64, 176, 64, 176, 1, 1, 1);
    // layer2 @ 32x88, 128ch (specs 5,6,7=down, 8,9)
    conv(5, M0, nullptr, M1, 64, 176, 32, 88, 2, 1, 1);
    conv(7, M0, nullptr, M3, 64, 176, 32, 88, 2, 0, 0);                 // downsample 1x1/2
    conv(6, M1, M3,      M2, 32, 88, 32, 88, 1, 1, 1);
    conv(8, M2, nullptr, M1, 32, 88, 32, 88, 1, 1, 1);
    conv(9, M1, M2,      M0, 32, 88, 32, 88, 1, 1, 1);
    // layer3 @ 16x44, 256ch (specs 10,11,12=down, 13,14)
    conv(10, M0, nullptr, M1, 32, 88, 16, 44, 2, 1, 1);
    conv(12, M0, nullptr, M3, 32, 88, 16, 44, 2, 0, 0);
    conv(11, M1, M3,      M2, 16, 44, 16, 44, 1, 1, 1);
    conv(13, M2, nullptr, M1, 16, 44, 16, 44, 1, 1, 1);
    conv(14, M1, M2,      M0, 16, 44, 16, 44, 1, 1, 1);
    // layer4 @ 8x22, 512ch (specs 15,16,17=down, 18,19)
    conv(15, M0, nullptr, M1, 16, 44, 8, 22, 2, 1, 1);
    conv(17, M0, nullptr, M3, 16, 44, 8, 22, 2, 0, 0);
    conv(16, M1, M3,      M2, 8, 22, 8, 22, 1, 1, 1);
    conv(18, M2, nullptr, M1, 8, 22, 8, 22, 1, 1, 1);
    conv(19, M1, M2,      M0, 8, 22, 8, 22, 1, 1, 1);
    // head 1x1 512->64 (+bias+bn+relu), spec 20
    conv(20, M0, nullptr, M1, 8, 22, 8, 22, 1, 0, 1);

    // geometry + scatter
    geom_kernel<<<cdiv(NPTS, 128), 128, 0, stream>>>(
        cam2ego, cam_ego2global, intrinsics, lidar2ego, ego2global,
        LIN, WMK, NC, 8, 22, 704.f / 22.f, 256.f / 8.f);
    zero_f32<<<cdiv((long)out_size, 256), 256, 0, stream>>>((float*)d_out, (size_t)out_size);
    scatter_bev<<<cdiv((long)NPTS * 64, 256), 256, 0, stream>>>(
        M1, LIN, WMK, (float*)d_out, NPTS, 64);
}
